// AttentionMemoryBank_20486994002595
// MI455X (gfx1250) — matrix operand order repaired and checked by reading, not compiled, not run on hardware
//
#include <hip/hip_runtime.h>
#include <hip/hip_fp16.h>

// Problem constants (match reference)
#define B_ROWS  2048
#define D_DIM   512
#define M_ROWS  32768
#define H_HEADS 4
#define HD_DIM  128
#define TOPK    64
#define CHUNK   128   // b-rows of scores materialized per pass (64MB f32)

typedef __attribute__((ext_vector_type(16))) _Float16 v16h;
typedef __attribute__((ext_vector_type(8)))  _Float16 v8h;
typedef __attribute__((ext_vector_type(8)))  float    v8f;

#if defined(__gfx1250__) && __has_builtin(__builtin_amdgcn_global_load_async_to_lds_b128)
#define HAVE_ASYNC_LDS 1
typedef int vi4 __attribute__((vector_size(16)));
typedef __attribute__((address_space(1))) vi4* as1_vi4p;
typedef __attribute__((address_space(3))) vi4* as3_vi4p;
#else
#define HAVE_ASYNC_LDS 0
#endif

// ---- fragment load helpers -------------------------------------------------
__device__ __forceinline__ v8h cvt8_f32(const float* p) {
    float4 x = *(const float4*)p;
    float4 y = *(const float4*)(p + 4);
    v8h r;
    r[0] = (_Float16)x.x; r[1] = (_Float16)x.y; r[2] = (_Float16)x.z; r[3] = (_Float16)x.w;
    r[4] = (_Float16)y.x; r[5] = (_Float16)y.y; r[6] = (_Float16)y.z; r[7] = (_Float16)y.w;
    return r;
}

__device__ __forceinline__ v16h cat16(v8h lo, v8h hi) {
    return __builtin_shufflevector(lo, hi, 0, 1, 2, 3, 4, 5, 6, 7,
                                           8, 9, 10, 11, 12, 13, 14, 15);
}

// A fragment (16x32 f16 layout): lane holds 8 halves at +0 and 8 halves at +16
template <bool IN_HALF>
__device__ __forceinline__ v16h load_afrag(const void* Av, long off) {
    if (IN_HALF) {
        const _Float16* p = (const _Float16*)Av + off;
        return cat16(*(const v8h*)p, *(const v8h*)(p + 16));
    } else {
        const float* p = (const float*)Av + off;
        return cat16(cvt8_f32(p), cvt8_f32(p + 16));
    }
}

// B fragment (32x16 f16 layout): lane holds 8 halves at +0 and 8 halves at +16
template <bool IN_HALF>
__device__ __forceinline__ v16h load_bfrag(const void* Bv, long off) {
    if (IN_HALF) {
        const _Float16* p = (const _Float16*)Bv + off;
        return cat16(*(const v8h*)p, *(const v8h*)(p + 16));
    } else {
        const float* p = (const float*)Bv + off;
        return cat16(cvt8_f32(p), cvt8_f32(p + 16));
    }
}

// ---------------------------------------------------------------------------
// Generic WMMA GEMM (projections / output):
//   C[i, j] = scale * dot(A[i, :KDIM], B[j, :KDIM]) + bias[j]
// Block = 128 threads (4 waves); each wave computes a 16(row) x 64(col) strip
// (4 accumulators sharing one A fragment per k-step).
// ---------------------------------------------------------------------------
template <int KDIM, bool IN_HALF, bool OUT_HALF>
__global__ __launch_bounds__(128)
void gemm_wmma_kernel(const void* __restrict__ Av, long lda,
                      const void* __restrict__ Bv, long ldb,
                      void* __restrict__ Cv, long cRowStride,
                      const float* __restrict__ bias, float scale)
{
    const int lane = threadIdx.x & 31;
    const int wave = threadIdx.x >> 5;
    const int rb   = blockIdx.y << 4;                 // row tile base
    const int cb   = ((blockIdx.x << 2) + wave) << 6; // 64-col strip per wave
    const int hi   = lane >> 4;
    const int l16  = lane & 15;

    const long aBase = (long)(rb + l16) * lda + hi * 8;

    long bBase[4];
    #pragma unroll
    for (int n = 0; n < 4; ++n)
        bBase[n] = (long)(cb + n * 16 + l16) * ldb + hi * 8;

    v8f acc[4] = {{}, {}, {}, {}};

    #pragma unroll 4
    for (int ks = 0; ks < KDIM; ks += 32) {
        if (IN_HALF)
            __builtin_prefetch((const _Float16*)Bv + bBase[0] + ks + 64, 0, 1);
        else
            __builtin_prefetch((const float*)Bv + bBase[0] + ks + 64, 0, 1);

        const v16h af = load_afrag<IN_HALF>(Av, aBase + ks);
        v16h bf[4];
        #pragma unroll
        for (int n = 0; n < 4; ++n)
            bf[n] = load_bfrag<IN_HALF>(Bv, bBase[n] + ks);

        #pragma unroll
        for (int n = 0; n < 4; ++n)
            acc[n] = __builtin_amdgcn_wmma_f32_16x16x32_f16(
                false, af, false, bf[n], (short)0, acc[n], false, false);
    }

    const int row0 = rb + hi * 8;
    #pragma unroll
    for (int n = 0; n < 4; ++n) {
        const int col = cb + n * 16 + l16;
        const float bv = bias ? bias[col] : 0.0f;
        #pragma unroll
        for (int v = 0; v < 8; ++v) {
            float val = acc[n][v] * scale + bv;
            long  idx = (long)(row0 + v) * cRowStride + col;
            if (OUT_HALF) ((_Float16*)Cv)[idx] = (_Float16)val;
            else          ((float*)Cv)[idx]    = val;
        }
    }
}

// ---------------------------------------------------------------------------
// Scores GEMM (dominant, 68.7 GF): scores[b, h, m] = scale * dot(Qh[b, h*128:],
// Kh[m, h*128:]).  Block = 128 threads (4 waves) stacked along rows; all waves
// share one 64-col x 128-K strip of Kh staged in LDS via async global->LDS
// copies (ASYNCcnt), padded to avoid bank conflicts. A fragments (K=128) are
// preloaded into registers. Inner loop: ds_load B frags + 16 WMMAs.
// Grid: (M/64, CHUNK/64, H).
// ---------------------------------------------------------------------------
#define BPAD 8
#define BSTRIDE (HD_DIM + BPAD)   // 136 halves per LDS row

__global__ __launch_bounds__(128)
void scores_wmma_kernel(const _Float16* __restrict__ Qc,   // CHUNK x 512
                        const _Float16* __restrict__ Kh,   // M x 512
                        float* __restrict__ scores, float scale)
{
    const int lane = threadIdx.x & 31;
    const int wave = threadIdx.x >> 5;
    const int tid  = threadIdx.x;
    const int cb   = blockIdx.x << 6;               // 64 cols (m) per block
    const int rb   = ((blockIdx.y << 2) + wave) << 4; // 16 rows per wave
    const int z    = blockIdx.z;                    // head
    const int hi   = lane >> 4;
    const int l16  = lane & 15;

    __shared__ _Float16 Bs[64 * BSTRIDE];           // 64 Kh rows x 128 K (+pad)

    // ---- stage Kh[cb .. cb+63, z*128 .. z*128+127] into LDS ----------------
    // 64 rows x 128 halves = 16KB -> 8 x 16B per thread
    #pragma unroll
    for (int i = 0; i < 8; ++i) {
        const int j  = tid + (i << 7);      // chunk id 0..1023
        const int r  = j >> 4;              // Kh row within strip
        const int k8 = (j & 15) << 3;       // half offset within row
        const _Float16* g = Kh + (long)(cb + r) * D_DIM + z * HD_DIM + k8;
        _Float16* l = &Bs[r * BSTRIDE + k8];
#if HAVE_ASYNC_LDS
        __builtin_amdgcn_global_load_async_to_lds_b128(
            (as1_vi4p)(g), (as3_vi4p)(l), 0, 0);
#else
        *(uint4*)l = *(const uint4*)g;
#endif
    }
#if HAVE_ASYNC_LDS
    asm volatile("s_wait_asynccnt 0x0" ::: "memory");
#endif
    __syncthreads();

    // ---- preload all 4 A fragments (K = 128) -------------------------------
    const long aBase = (long)(rb + l16) * D_DIM + z * HD_DIM + hi * 8;
    v16h af[4];
    #pragma unroll
    for (int t = 0; t < 4; ++t)
        af[t] = cat16(*(const v8h*)(Qc + aBase + t * 32),
                      *(const v8h*)(Qc + aBase + t * 32 + 16));

    // ---- 16 WMMAs from LDS-resident B --------------------------------------
    v8f acc[4] = {{}, {}, {}, {}};
    #pragma unroll
    for (int ks = 0; ks < 4; ++ks) {
        #pragma unroll
        for (int n = 0; n < 4; ++n) {
            const _Float16* bp = &Bs[(n * 16 + l16) * BSTRIDE + ks * 32 + hi * 8];
            v16h bf = cat16(*(const v8h*)bp, *(const v8h*)(bp + 16));
            acc[n] = __builtin_amdgcn_wmma_f32_16x16x32_f16(
                false, af[ks], false, bf, (short)0, acc[n], false, false);
        }
    }

    // ---- store scores chunk ------------------------------------------------
    const int row0 = rb + hi * 8;
    #pragma unroll
    for (int n = 0; n < 4; ++n) {
        const int col = cb + n * 16 + l16;
        #pragma unroll
        for (int v = 0; v < 8; ++v) {
            long idx = (long)(row0 + v) * (H_HEADS * M_ROWS) + (long)z * M_ROWS + col;
            scores[idx] = acc[n][v] * scale;
        }
    }
}

// ---------------------------------------------------------------------------
// Per-(b,h) exact top-64 over M=32768 scores (iterative block argmax with
// in-place masking), softmax over the 64 winners, gather of V rows and
// weighted sum into attended[b, h*HD : (h+1)*HD].
// Block = 256 threads; grid = (H, CHUNK).
// ---------------------------------------------------------------------------
__global__ __launch_bounds__(256)
void topk_softmax_gather_kernel(float* __restrict__ scores,
                                const _Float16* __restrict__ Vh,
                                float* __restrict__ attended, int bOffset)
{
    const int h   = blockIdx.x;
    const int bl  = blockIdx.y;
    const int tid = threadIdx.x;
    float* srow = scores + ((long)bl * H_HEADS + h) * M_ROWS;

    const int seg = M_ROWS / 256; // 128 elems per thread
    const int s0  = tid * seg;

    float lmax = -3.0e38f; int lidx = s0;
    for (int i = 0; i < seg; ++i) {
        float v = srow[s0 + i];
        if (v > lmax) { lmax = v; lidx = s0 + i; }
    }

    __shared__ float sv[256];
    __shared__ int   si[256];
    __shared__ float topv[TOPK];
    __shared__ int   topi[TOPK];

    for (int it = 0; it < TOPK; ++it) {
        sv[tid] = lmax; si[tid] = lidx;
        __syncthreads();
        for (int off = 128; off > 0; off >>= 1) {
            if (tid < off && sv[tid + off] > sv[tid]) {
                sv[tid] = sv[tid + off]; si[tid] = si[tid + off];
            }
            __syncthreads();
        }
        const int widx = si[0];
        if (tid == 0) { topv[it] = sv[0]; topi[it] = widx; }
        if (widx >= s0 && widx < s0 + seg) {
            srow[widx] = -3.0e38f;           // mask winner, rescan own segment
            lmax = -3.0e38f; lidx = s0;
            for (int i = 0; i < seg; ++i) {
                float v = srow[s0 + i];
                if (v > lmax) { lmax = v; lidx = s0 + i; }
            }
        }
        __syncthreads();
    }

    // softmax over 64 (topv is sorted descending -> topv[0] is the max)
    __shared__ float wts[TOPK];
    __shared__ float sumsh;
    if (tid < TOPK) wts[tid] = __expf(topv[tid] - topv[0]);
    __syncthreads();
    if (tid == 0) {
        float s = 0.0f;
        for (int i = 0; i < TOPK; ++i) s += wts[i];
        sumsh = s;
    }
    __syncthreads();

    // gather + weighted sum: one thread per output feature d in [0, HD)
    if (tid < HD_DIM) {
        const _Float16* vb = Vh + (long)h * HD_DIM + tid;
        float a = 0.0f;
        for (int t = 0; t < TOPK; ++t)
            a += wts[t] * (float)vb[(long)topi[t] * D_DIM];
        attended[(long)(bOffset + bl) * D_DIM + h * HD_DIM + tid] = a / sumsh;
    }
}

// ---------------------------------------------------------------------------
extern "C" void kernel_launch(void* const* d_in, const int* in_sizes, int n_in,
                              void* d_out, int out_size, void* d_ws, size_t ws_size,
                              hipStream_t stream)
{
    (void)in_sizes; (void)n_in; (void)out_size; (void)ws_size;

    const float* query  = (const float*)d_in[0];
    const float* memory = (const float*)d_in[1];
    const float* Wq = (const float*)d_in[2];
    const float* bq = (const float*)d_in[3];
    const float* Wk = (const float*)d_in[4];
    const float* bk = (const float*)d_in[5];
    const float* Wv = (const float*)d_in[6];
    const float* bv = (const float*)d_in[7];
    const float* Wo = (const float*)d_in[8];
    const float* bo = (const float*)d_in[9];
    float* out = (float*)d_out;

    // Workspace layout (~134 MB total)
    char* ws = (char*)d_ws;
    _Float16* Qh = (_Float16*)ws;  ws += (size_t)B_ROWS * D_DIM * sizeof(_Float16);
    _Float16* Kh = (_Float16*)ws;  ws += (size_t)M_ROWS * D_DIM * sizeof(_Float16);
    _Float16* Vh = (_Float16*)ws;  ws += (size_t)M_ROWS * D_DIM * sizeof(_Float16);
    float* attended = (float*)ws;  ws += (size_t)B_ROWS * D_DIM * sizeof(float);
    float* scores   = (float*)ws;  // CHUNK * H * M * 4 bytes

    const dim3 blk(128);

    // Projections: X @ W.T + b  (f32 in, f16 out). Block tile = 16 x 256.
    gemm_wmma_kernel<D_DIM, false, true><<<dim3(2, B_ROWS / 16, 1), blk, 0, stream>>>(
        query, D_DIM, Wq, D_DIM, Qh, D_DIM, bq, 1.0f);
    gemm_wmma_kernel<D_DIM, false, true><<<dim3(2, M_ROWS / 16, 1), blk, 0, stream>>>(
        memory, D_DIM, Wk, D_DIM, Kh, D_DIM, bk, 1.0f);
    gemm_wmma_kernel<D_DIM, false, true><<<dim3(2, M_ROWS / 16, 1), blk, 0, stream>>>(
        memory, D_DIM, Wv, D_DIM, Vh, D_DIM, bv, 1.0f);

    // Scores + top-k + softmax + gather, chunked over B rows
    const float scale = 0.08838834764831845f; // 1/sqrt(HD)
    for (int c = 0; c < B_ROWS / CHUNK; ++c) {
        const _Float16* Qc = Qh + (size_t)c * CHUNK * D_DIM;
        scores_wmma_kernel<<<dim3(M_ROWS / 64, CHUNK / 64, H_HEADS), blk, 0, stream>>>(
            Qc, Kh, scores, scale);
        topk_softmax_gather_kernel<<<dim3(H_HEADS, CHUNK), 256, 0, stream>>>(
            scores, Vh, attended, c * CHUNK);
    }

    // Output projection: attended @ Wo.T + bo  (f32 in, f32 out)
    gemm_wmma_kernel<D_DIM, false, false><<<dim3(2, B_ROWS / 16, 1), blk, 0, stream>>>(
        attended, D_DIM, Wo, D_DIM, out, D_DIM, bo, 1.0f);
}
